// DIEN_61366492725611
// MI455X (gfx1250) — matrix operand order repaired and checked by reading, not compiled, not run on hardware
//
#include <hip/hip_runtime.h>
#include <math.h>

typedef __attribute__((ext_vector_type(16))) _Float16 v16h;
typedef __attribute__((ext_vector_type(8)))  float    v8f;

#define B_  512
#define T_  200
#define D_  128
#define BT_ (B_ * T_)

__device__ __forceinline__ float sigmf(float x) { return 1.0f / (1.0f + __expf(-x)); }

// ---- WMMA tile helpers (CDNA5 16x16x32 f16, layouts per CDNA5 ISA) ----

// A tile: 16 rows x 32 K, row-major f16 source with leading dim ld (halves).
// lane<16: row=lane,   K = {0..7, 16..23}
// lane>=16: row=lane-16, K = {8..15, 24..31}
__device__ __forceinline__ v16h load_a16(const _Float16* base, int ld) {
  int lane = threadIdx.x & 31;
  const _Float16* p = base + (size_t)(lane & 15) * (size_t)ld + ((lane < 16) ? 0 : 8);
  v16h a;
#pragma unroll
  for (int i = 0; i < 8; ++i) a[i] = p[i];
#pragma unroll
  for (int i = 0; i < 8; ++i) a[i + 8] = p[16 + i];
  return a;
}

// B tile for X @ W^T: source W row-major (N,K), ld = K stride (halves).
// B[k][n] = W[n][k]; lane n = lane%16; lane<16 -> K={0..7,16..23}, lane>=16 -> K={8..15,24..31}.
__device__ __forceinline__ v16h load_bt16(const _Float16* w, int ld) {
  int lane = threadIdx.x & 31;
  const _Float16* p = w + (size_t)(lane & 15) * (size_t)ld + ((lane < 16) ? 0 : 8);
  v16h b;
#pragma unroll
  for (int i = 0; i < 16; ++i) b[i] = p[(i < 8) ? i : (i + 8)];
  return b;
}

__device__ __forceinline__ v8f wmma16(v16h a, v16h b, v8f c) {
  return __builtin_amdgcn_wmma_f32_16x16x32_f16(false, a, false, b, (short)0, c,
                                                false, false);
}

// ---------------- prep kernels ----------------

__global__ void k_f2h(_Float16* __restrict__ dst, const float* __restrict__ src, int n) {
  for (int i = blockIdx.x * 256 + threadIdx.x; i < n; i += gridDim.x * 256)
    dst[i] = (_Float16)src[i];
}
__global__ void k_zero_f32(float* __restrict__ d, int n) {
  for (int i = blockIdx.x * 256 + threadIdx.x; i < n; i += gridDim.x * 256) d[i] = 0.0f;
}
__global__ void k_zero_f16(_Float16* __restrict__ d, int n) {
  for (int i = blockIdx.x * 256 + threadIdx.x; i < n; i += gridDim.x * 256)
    d[i] = (_Float16)0.0f;
}
// attn_W0 (80,512) -> padded (96,512)
__global__ void k_padW0(_Float16* __restrict__ dst, const float* __restrict__ src) {
  int i = blockIdx.x * 256 + threadIdx.x;
  if (i >= 96 * 512) return;
  int r = i >> 9, c = i & 511;
  dst[i] = (r < 80) ? (_Float16)src[r * 512 + c] : (_Float16)0.0f;
}
// attn_W1 (40,80) -> padded (48,96)
__global__ void k_padW1(_Float16* __restrict__ dst, const float* __restrict__ src) {
  int i = blockIdx.x * 256 + threadIdx.x;
  if (i >= 48 * 96) return;
  int r = i / 96, c = i % 96;
  dst[i] = (r < 40 && c < 80) ? (_Float16)src[r * 80 + c] : (_Float16)0.0f;
}

// ---------------- history mean ----------------
__global__ void k_hist(float* __restrict__ hist, const float* __restrict__ X,
                       const int* __restrict__ seqlen) {
  int idx = blockIdx.x * 256 + threadIdx.x;  // B*D = 65536
  int b = idx >> 7, d = idx & 127;
  int len = seqlen[b];
  const float* p = X + (size_t)b * T_ * D_ + d;
  float s = 0.0f;
  for (int t = 0; t < len; ++t) s += p[(size_t)t * D_];
  hist[idx] = s / (float)len;
}

// ---------------- gi = X @ Wih^T + bih  (M=BT, N=384, K=128) ----------------
__global__ void __launch_bounds__(256)
k_gemm_gi(float* __restrict__ gi, const _Float16* __restrict__ X16,
          const _Float16* __restrict__ Wih16, const float* __restrict__ bih) {
  const int m0 = blockIdx.x * 16;
  const int w = threadIdx.x >> 5;
  const int lane = threadIdx.x & 31;
  v16h a[4];
#pragma unroll
  for (int kc = 0; kc < 4; ++kc)
    a[kc] = load_a16(X16 + (size_t)m0 * D_ + kc * 32, D_);
#pragma unroll
  for (int i = 0; i < 3; ++i) {
    int n0 = (w * 3 + i) * 16;
    v8f acc = {};
#pragma unroll
    for (int kc = 0; kc < 4; ++kc) {
      v16h b = load_bt16(Wih16 + (size_t)n0 * D_ + kc * 32, D_);
      acc = wmma16(a[kc], b, acc);
    }
    int n = n0 + (lane & 15);
    int rb = (lane < 16) ? 0 : 8;
    float bv = bih[n];
#pragma unroll
    for (int g = 0; g < 8; ++g)
      gi[(size_t)(m0 + rb + g) * 384 + n] = acc[g] + bv;
  }
}

// ---------------- one GRU step: gh = h @ Whh^T (+bhh), gates, new h ----------------
__global__ void __launch_bounds__(256)
k_gru_step(int t, const float* __restrict__ gi, const float* __restrict__ bhh,
           const _Float16* __restrict__ Whh16, float* __restrict__ h32,
           _Float16* __restrict__ h16, _Float16* __restrict__ gru16) {
  __shared__ float gh[16 * 384];
  const int b0 = blockIdx.x * 16;
  const int w = threadIdx.x >> 5;
  const int lane = threadIdx.x & 31;

  v16h a[4];
#pragma unroll
  for (int kc = 0; kc < 4; ++kc)
    a[kc] = load_a16(h16 + (size_t)b0 * D_ + kc * 32, D_);

#pragma unroll
  for (int i = 0; i < 3; ++i) {
    int n0 = (w * 3 + i) * 16;
    v8f acc = {};
#pragma unroll
    for (int kc = 0; kc < 4; ++kc) {
      v16h b = load_bt16(Whh16 + (size_t)n0 * D_ + kc * 32, D_);
      acc = wmma16(a[kc], b, acc);
    }
    int n = n0 + (lane & 15);
    int rb = (lane < 16) ? 0 : 8;
#pragma unroll
    for (int g = 0; g < 8; ++g) gh[(rb + g) * 384 + n] = acc[g];
  }
  __syncthreads();

#pragma unroll
  for (int it = 0; it < 8; ++it) {
    int idx = it * 256 + threadIdx.x;
    int row = idx >> 7, col = idx & 127;
    int b = b0 + row;
    const float* gir = gi + ((size_t)b * T_ + t) * 384;
    float r = sigmf(gir[col] + gh[row * 384 + col] + bhh[col]);
    float z = sigmf(gir[128 + col] + gh[row * 384 + 128 + col] + bhh[128 + col]);
    float nn = tanhf(gir[256 + col] + r * (gh[row * 384 + 256 + col] + bhh[256 + col]));
    float hp = h32[(size_t)b * D_ + col];
    float hn = (1.0f - z) * nn + z * hp;
    h32[(size_t)b * D_ + col] = hn;
    h16[(size_t)b * D_ + col] = (_Float16)hn;
    gru16[((size_t)b * T_ + t) * D_ + col] = (_Float16)hn;
  }
}

// ------- attention layer0: x=[fact,q,fact*q,q-fact] (K=512) -> relu(@W0^T+b0), N=96(pad) -------
__global__ void __launch_bounds__(256)
k_attn0(_Float16* __restrict__ y0, const _Float16* __restrict__ gru16,
        const _Float16* __restrict__ q16, const _Float16* __restrict__ W0p,
        const float* __restrict__ b0v) {
  const int w = threadIdx.x >> 5;
  if (w >= 6) return;  // 6 N-tiles; wave-uniform exit, EXEC stays all-ones in live waves
  const int lane = threadIdx.x & 31;
  const int m0 = blockIdx.x * 16;
  const int row = m0 + (lane & 15);
  const int b = row / T_;
  const _Float16* fr = gru16 + (size_t)row * D_;
  const _Float16* qr = q16 + (size_t)b * D_;
  const int kb = (lane < 16) ? 0 : 8;
  const int n0 = w * 16;

  v8f acc = {};
#pragma unroll
  for (int k = 0; k < 16; ++k) {
    int seg = k >> 2;            // each 32-wide K chunk sits inside one 128-wide segment
    int base = (k * 32) & 127;
    v16h a;
#pragma unroll
    for (int i = 0; i < 16; ++i) {
      int off = (i < 8) ? (kb + i) : (kb + 8 + i);
      int j = base + off;
      float f = (float)fr[j];
      float q = (float)qr[j];
      float v = (seg == 0) ? f : (seg == 1) ? q : (seg == 2) ? f * q : (q - f);
      a[i] = (_Float16)v;
    }
    v16h bm = load_bt16(W0p + (size_t)n0 * 512 + k * 32, 512);
    acc = wmma16(a, bm, acc);
  }
  int n = n0 + (lane & 15);
  int rb = (lane < 16) ? 0 : 8;
  float bv = (n < 80) ? b0v[n] : 0.0f;
#pragma unroll
  for (int g = 0; g < 8; ++g)
    y0[(size_t)(m0 + rb + g) * 96 + n] = (_Float16)fmaxf(acc[g] + bv, 0.0f);
}

// ------- attention layer1: relu(y0 @ W1p^T + b1), M=BT, N=48(pad), K=96(pad) -------
__global__ void __launch_bounds__(256)
k_attn1(_Float16* __restrict__ y1, const _Float16* __restrict__ y0,
        const _Float16* __restrict__ W1p, const float* __restrict__ b1v) {
  const int w = threadIdx.x >> 5;
  if (w >= 3) return;
  const int lane = threadIdx.x & 31;
  const int m0 = blockIdx.x * 16;
  const int n0 = w * 16;
  v8f acc = {};
#pragma unroll
  for (int k = 0; k < 3; ++k) {
    v16h a = load_a16(y0 + (size_t)m0 * 96 + k * 32, 96);
    v16h b = load_bt16(W1p + (size_t)n0 * 96 + k * 32, 96);
    acc = wmma16(a, b, acc);
  }
  int n = n0 + (lane & 15);
  int rb = (lane < 16) ? 0 : 8;
  float bv = (n < 40) ? b1v[n] : 0.0f;
#pragma unroll
  for (int g = 0; g < 8; ++g)
    y1[(size_t)(m0 + rb + g) * 48 + n] = (_Float16)fmaxf(acc[g] + bv, 0.0f);
}

// ------- layer2 (40->1) + masked softmax over T per batch row -------
__global__ void __launch_bounds__(256)
k_attn2(float* __restrict__ sc, const _Float16* __restrict__ y1,
        const float* __restrict__ W2, const float* __restrict__ b2,
        const int* __restrict__ seqlen) {
  __shared__ float red[256];
  const int b = blockIdx.x;
  const int t = threadIdx.x;
  const int len = seqlen[b];
  float raw = 0.0f;
  bool valid = (t < T_) && (t < len);
  if (t < T_) {
    const _Float16* yr = y1 + ((size_t)b * T_ + t) * 48;
    float s = b2[0];
#pragma unroll
    for (int j = 0; j < 40; ++j) s += (float)yr[j] * W2[j];
    raw = fmaxf(s, 0.0f);
  }
  red[t] = valid ? raw : -3.4e38f;
  __syncthreads();
  for (int s = 128; s > 0; s >>= 1) {
    if (t < s) red[t] = fmaxf(red[t], red[t + s]);
    __syncthreads();
  }
  float mx = red[0];
  __syncthreads();
  float e = valid ? __expf(raw - mx) : 0.0f;
  red[t] = e;
  __syncthreads();
  for (int s = 128; s > 0; s >>= 1) {
    if (t < s) red[t] += red[t + s];
    __syncthreads();
  }
  float sum = red[0];
  if (t < T_) sc[(size_t)b * T_ + t] = valid ? (e / sum) : 0.0f;
}

// ---------------- one AUGRU step ----------------
__global__ void __launch_bounds__(256)
k_augru_step(int t, const float* __restrict__ sc, const _Float16* __restrict__ gru16,
             const _Float16* __restrict__ Wr16, const _Float16* __restrict__ Wu16,
             const _Float16* __restrict__ Wh16, const float* __restrict__ br,
             const float* __restrict__ bu, const float* __restrict__ bh,
             const int* __restrict__ seqlen, float* __restrict__ ah32,
             _Float16* __restrict__ ah16, float* __restrict__ attf) {
  __shared__ float rl[16 * 128];
  __shared__ float ul[16 * 128];
  __shared__ _Float16 hrl[16 * 128];
  __shared__ float hhl[16 * 128];

  const int b0 = blockIdx.x * 16;
  const int w = threadIdx.x >> 5;
  const int lane = threadIdx.x & 31;

  // A = [h(K 0..127), x(K 128..255)]; x rows strided by T*D
  v16h a[8];
#pragma unroll
  for (int kc = 0; kc < 4; ++kc)
    a[kc] = load_a16(ah16 + (size_t)b0 * D_ + kc * 32, D_);
#pragma unroll
  for (int kc = 0; kc < 4; ++kc)
    a[4 + kc] = load_a16(gru16 + ((size_t)b0 * T_ + t) * D_ + kc * 32, T_ * D_);

  // phase 1: r (waves 0-3) and u (waves 4-7); 2 N-tiles per wave, K=256
  {
    const _Float16* W = (w < 4) ? Wr16 : Wu16;
    const float* bias = (w < 4) ? br : bu;
    float* dst = (w < 4) ? rl : ul;
    int wl = w & 3;
#pragma unroll
    for (int i = 0; i < 2; ++i) {
      int n0 = (wl * 2 + i) * 16;
      v8f acc = {};
#pragma unroll
      for (int kc = 0; kc < 8; ++kc) {
        v16h b = load_bt16(W + (size_t)n0 * 256 + kc * 32, 256);
        acc = wmma16(a[kc], b, acc);
      }
      int n = n0 + (lane & 15);
      int rb = (lane < 16) ? 0 : 8;
      float bv = bias[n];
#pragma unroll
      for (int g = 0; g < 8; ++g) dst[(rb + g) * 128 + n] = sigmf(acc[g] + bv);
    }
  }
  __syncthreads();
  // phase 2: hr = h * r (f16) into LDS
#pragma unroll
  for (int it = 0; it < 8; ++it) {
    int idx = it * 256 + threadIdx.x;
    int row = idx >> 7, col = idx & 127;
    hrl[idx] = (_Float16)(rl[idx] * ah32[(size_t)(b0 + row) * D_ + col]);
  }
  __syncthreads();
  // phase 3: h_hat = tanh([h*r, x] @ Wh^T + bh); 1 N-tile per wave
  {
    int n0 = w * 16;
    v8f acc = {};
#pragma unroll
    for (int kc = 0; kc < 4; ++kc) {
      v16h ahr = load_a16(hrl + kc * 32, 128);  // LDS-sourced A tile
      v16h b = load_bt16(Wh16 + (size_t)n0 * 256 + kc * 32, 256);
      acc = wmma16(ahr, b, acc);
    }
#pragma unroll
    for (int kc = 0; kc < 4; ++kc) {
      v16h b = load_bt16(Wh16 + (size_t)n0 * 256 + (4 + kc) * 32, 256);
      acc = wmma16(a[4 + kc], b, acc);
    }
    int n = n0 + (lane & 15);
    int rb = (lane < 16) ? 0 : 8;
    float bv = bh[n];
#pragma unroll
    for (int g = 0; g < 8; ++g) hhl[(rb + g) * 128 + n] = tanhf(acc[g] + bv);
  }
  __syncthreads();
  // phase 4: h = (1 - a*u)*h + a*u*h_hat; capture attention_feature at t==len-1
#pragma unroll
  for (int it = 0; it < 8; ++it) {
    int idx = it * 256 + threadIdx.x;
    int row = idx >> 7, col = idx & 127;
    int b = b0 + row;
    float aw = sc[(size_t)b * T_ + t];
    float u = aw * ul[idx];
    float hp = ah32[(size_t)b * D_ + col];
    float hn = (1.0f - u) * hp + u * hhl[idx];
    ah32[(size_t)b * D_ + col] = hn;
    ah16[(size_t)b * D_ + col] = (_Float16)hn;
    if (t == seqlen[b] - 1) attf[(size_t)b * D_ + col] = hn;
  }
}

// ---------------- final 640 -> 1 ----------------
__global__ void k_final(float* __restrict__ out, const float* __restrict__ user,
                        const float* __restrict__ item, const float* __restrict__ hist,
                        const float* __restrict__ attf, const float* __restrict__ Wo,
                        const float* __restrict__ bo) {
  __shared__ float red[256];
  const int b = blockIdx.x;
  const int t = threadIdx.x;
  float s = 0.0f;
  for (int idx = t; idx < 640; idx += 256) {
    int seg = idx >> 7, j = idx & 127;
    float v;
    if (seg == 0) v = user[(size_t)b * 128 + j];
    else if (seg == 1) v = item[(size_t)b * 128 + j];
    else if (seg == 2) v = hist[(size_t)b * 128 + j];
    else if (seg == 3) v = item[(size_t)b * 128 + j] * hist[(size_t)b * 128 + j];
    else v = attf[(size_t)b * 128 + j];
    s += v * Wo[idx];
  }
  red[t] = s;
  __syncthreads();
  for (int st = 128; st > 0; st >>= 1) {
    if (t < st) red[t] += red[t + st];
    __syncthreads();
  }
  if (t == 0) out[b] = red[0] + bo[0];
}

// ---------------- host launcher ----------------
extern "C" void kernel_launch(void* const* d_in, const int* in_sizes, int n_in,
                              void* d_out, int out_size, void* d_ws, size_t ws_size,
                              hipStream_t stream) {
  const float* user = (const float*)d_in[0];
  const float* X = (const float*)d_in[1];
  const float* item = (const float*)d_in[2];
  const int* seq = (const int*)d_in[3];
  // d_in[4] = mask (recomputed from seq on device)
  const float* Wih = (const float*)d_in[5];
  const float* Whh = (const float*)d_in[6];
  const float* bih = (const float*)d_in[7];
  const float* bhh = (const float*)d_in[8];
  const float* aW0 = (const float*)d_in[9];
  const float* ab0 = (const float*)d_in[10];
  const float* aW1 = (const float*)d_in[11];
  const float* ab1 = (const float*)d_in[12];
  const float* aW2 = (const float*)d_in[13];
  const float* ab2 = (const float*)d_in[14];
  const float* Wr = (const float*)d_in[15];
  const float* Wu = (const float*)d_in[16];
  const float* Wh = (const float*)d_in[17];
  const float* br = (const float*)d_in[18];
  const float* bu = (const float*)d_in[19];
  const float* bh = (const float*)d_in[20];
  const float* Wo = (const float*)d_in[21];
  const float* bo = (const float*)d_in[22];
  float* out = (float*)d_out;
  (void)in_sizes; (void)n_in; (void)out_size; (void)ws_size;

  char* wsb = (char*)d_ws;
  size_t off = 0;
  auto alloc = [&](size_t bytes) -> void* {
    void* p = wsb + off;
    off = (off + bytes + 255) & ~(size_t)255;
    return p;
  };
  _Float16* X16 = (_Float16*)alloc((size_t)BT_ * D_ * 2);
  _Float16* Wih16 = (_Float16*)alloc(384 * 128 * 2);
  _Float16* Whh16 = (_Float16*)alloc(384 * 128 * 2);
  _Float16* Wr16 = (_Float16*)alloc(128 * 256 * 2);
  _Float16* Wu16 = (_Float16*)alloc(128 * 256 * 2);
  _Float16* Wh16 = (_Float16*)alloc(128 * 256 * 2);
  _Float16* W0p = (_Float16*)alloc(96 * 512 * 2);
  _Float16* W1p = (_Float16*)alloc(48 * 96 * 2);
  _Float16* q16 = (_Float16*)alloc((size_t)B_ * D_ * 2);
  float* gi = (float*)alloc((size_t)BT_ * 384 * 4);
  float* h32 = (float*)alloc((size_t)B_ * D_ * 4);
  _Float16* h16 = (_Float16*)alloc((size_t)B_ * D_ * 2);
  _Float16* gru16 = (_Float16*)alloc((size_t)BT_ * D_ * 2);
  float* hist = (float*)alloc((size_t)B_ * D_ * 4);
  _Float16* y0 = (_Float16*)alloc((size_t)BT_ * 96 * 2);
  _Float16* y1 = (_Float16*)alloc((size_t)BT_ * 48 * 2);
  float* sc = (float*)alloc((size_t)BT_ * 4);
  float* ah32 = (float*)alloc((size_t)B_ * D_ * 4);
  _Float16* ah16 = (_Float16*)alloc((size_t)B_ * D_ * 2);
  float* attf = (float*)alloc((size_t)B_ * D_ * 4);

  dim3 blk(256);
  k_f2h<<<4096, blk, 0, stream>>>(X16, X, BT_ * D_);
  k_f2h<<<192, blk, 0, stream>>>(Wih16, Wih, 384 * 128);
  k_f2h<<<192, blk, 0, stream>>>(Whh16, Whh, 384 * 128);
  k_f2h<<<128, blk, 0, stream>>>(Wr16, Wr, 128 * 256);
  k_f2h<<<128, blk, 0, stream>>>(Wu16, Wu, 128 * 256);
  k_f2h<<<128, blk, 0, stream>>>(Wh16, Wh, 128 * 256);
  k_f2h<<<256, blk, 0, stream>>>(q16, item, B_ * D_);
  k_padW0<<<192, blk, 0, stream>>>(W0p, aW0);
  k_padW1<<<18, blk, 0, stream>>>(W1p, aW1);
  k_zero_f32<<<256, blk, 0, stream>>>(h32, B_ * D_);
  k_zero_f16<<<256, blk, 0, stream>>>(h16, B_ * D_);
  k_zero_f32<<<256, blk, 0, stream>>>(ah32, B_ * D_);
  k_zero_f16<<<256, blk, 0, stream>>>(ah16, B_ * D_);
  k_zero_f32<<<256, blk, 0, stream>>>(attf, B_ * D_);

  k_hist<<<256, blk, 0, stream>>>(hist, X, seq);
  k_gemm_gi<<<BT_ / 16, blk, 0, stream>>>(gi, X16, Wih16, bih);
  for (int t = 0; t < T_; ++t)
    k_gru_step<<<B_ / 16, blk, 0, stream>>>(t, gi, bhh, Whh16, h32, h16, gru16);
  k_attn0<<<BT_ / 16, blk, 0, stream>>>(y0, gru16, q16, W0p, ab0);
  k_attn1<<<BT_ / 16, blk, 0, stream>>>(y1, y0, W1p, ab1);
  k_attn2<<<B_, blk, 0, stream>>>(sc, y1, aW2, ab2, seq);
  for (int t = 0; t < T_; ++t)
    k_augru_step<<<B_ / 16, blk, 0, stream>>>(t, sc, gru16, Wr16, Wu16, Wh16, br, bu,
                                              bh, seq, ah32, ah16, attf);
  k_final<<<B_, blk, 0, stream>>>(out, user, item, hist, attf, Wo, bo);
}
